// _BnbMultiheadAttention_51058571214915
// MI455X (gfx1250) — compile-verified, matrix operand order checked by reading, not run on hardware
//
#include <hip/hip_runtime.h>

// ---------------------------------------------------------------------------
// MI455X / gfx1250 multi-head self-attention (B=2,S=2048,E=1024,H=16,D=64)
//   k1: in_proj  (fp32 -> bf16 WMMA; Q,K as [B,H,S,D], V pre-transposed [B,H,D,S])
//   k2: flash attention, 64-key blocks, double-buffered async K/V staging
//   k3: out_proj (async-staged bf16 A, fp32 out + bias)
// Matmuls: V_WMMA_F32_16X16X32_BF16 (wave32). Async staging:
// GLOBAL_LOAD_ASYNC_TO_LDS_B128 + S_WAIT_ASYNCCNT (inline asm, ASYNCcnt).
// ---------------------------------------------------------------------------

typedef unsigned short u16;
typedef __attribute__((ext_vector_type(4)))  float  f32x4;
typedef __attribute__((ext_vector_type(8)))  float  f32x8;
typedef __attribute__((ext_vector_type(4)))  u16    u16x4;
typedef __attribute__((ext_vector_type(8)))  u16    u16x8;
typedef __attribute__((ext_vector_type(16))) u16    u16x16;
typedef __attribute__((ext_vector_type(16))) __bf16 bf16x16;

#define B_DIM 2
#define S_DIM 2048
#define E_DIM 1024
#define H_DIM 16
#define D_DIM 64

// fp32 -> bf16 (round to nearest even)
__device__ __forceinline__ u16 f2bf(float f) {
  unsigned u = __builtin_bit_cast(unsigned, f);
  u += 0x7FFFu + ((u >> 16) & 1u);
  return (u16)(u >> 16);
}

// 16x32 bf16 A/B fragment row load for this lane (ISA 7.12.2 packing):
//   elems 0..7  -> K = kh*8 + 0..7 ; elems 8..15 -> K = 16 + kh*8 + 0..7
__device__ __forceinline__ bf16x16 frag_ld(const u16* row, int kh) {
  u16x8 lo = *(const u16x8*)(row + kh * 8);
  u16x8 hi = *(const u16x8*)(row + 16 + kh * 8);
  u16x16 v = __builtin_shufflevector(lo, hi, 0, 1, 2, 3, 4, 5, 6, 7,
                                     8, 9, 10, 11, 12, 13, 14, 15);
  return __builtin_bit_cast(bf16x16, v);
}

__device__ __forceinline__ f32x8 wmma_bf16(bf16x16 a, bf16x16 b, f32x8 c) {
  return __builtin_amdgcn_wmma_f32_16x16x32_bf16(false, a, false, b,
                                                 (short)0, c, false, false);
}

// Async 16B copy global -> LDS (ASYNCcnt). Generic LDS pointers carry the
// LDS byte offset in their low 32 bits (aperture form) -> VDST operand.
__device__ __forceinline__ void async_b128(u16* lds, const u16* gaddr) {
  unsigned off = (unsigned)(size_t)lds;
  asm volatile("global_load_async_to_lds_b128 %0, %1, off" ::"v"(off),
               "v"(gaddr)
               : "memory");
}
__device__ __forceinline__ void wait_async0() {
  asm volatile("s_wait_asynccnt 0x0" ::: "memory");
}

// ---------------------------------------------------------------------------
// Kernel 1: proj = X @ Wqkv^T + bias; Q,K -> [B,H,S,D]; V -> [B,H,D,S] (pre-T)
// Grid: (4096/128, 3072/64); 256 threads (8 waves: 4 along M x 2 along N)
// ---------------------------------------------------------------------------
__global__ __launch_bounds__(256) void k_inproj(
    const float* __restrict__ X, const float* __restrict__ W,
    const float* __restrict__ bias, u16* __restrict__ Qb,
    u16* __restrict__ Kb, u16* __restrict__ Vb) {
  __shared__ u16 As[128][40];  // stride 40 u16 = 80B: 16B-aligned, 0-conflict
  __shared__ u16 Bs[64][40];

  const int tid = threadIdx.x, lane = tid & 31, wid = tid >> 5;
  const int wm = wid & 3, wn = wid >> 2;
  const int kh = lane >> 4, ln = lane & 15;
  const int mB = blockIdx.x * 128, nB = blockIdx.y * 64;

  f32x8 acc[2][2];
#pragma unroll
  for (int i = 0; i < 2; i++)
#pragma unroll
    for (int j = 0; j < 2; j++)
#pragma unroll
      for (int v = 0; v < 8; v++) acc[i][j][v] = 0.f;

  for (int k0 = 0; k0 < E_DIM; k0 += 32) {
#pragma unroll
    for (int it = 0; it < 4; ++it) {  // A: 128x32 fp32 -> bf16
      int slot = it * 256 + tid;
      int r = slot >> 3, c4 = slot & 7;
      f32x4 xv = *(const f32x4*)(X + (size_t)(mB + r) * E_DIM + k0 + c4 * 4);
      u16x4 bv;
      bv[0] = f2bf(xv[0]); bv[1] = f2bf(xv[1]);
      bv[2] = f2bf(xv[2]); bv[3] = f2bf(xv[3]);
      *(u16x4*)(&As[r][c4 * 4]) = bv;
    }
#pragma unroll
    for (int it = 0; it < 2; ++it) {  // B: 64 rows of Wqkv
      int slot = it * 256 + tid;
      int r = slot >> 3, c4 = slot & 7;
      f32x4 wv = *(const f32x4*)(W + (size_t)(nB + r) * E_DIM + k0 + c4 * 4);
      u16x4 bv;
      bv[0] = f2bf(wv[0]); bv[1] = f2bf(wv[1]);
      bv[2] = f2bf(wv[2]); bv[3] = f2bf(wv[3]);
      *(u16x4*)(&Bs[r][c4 * 4]) = bv;
    }
    __syncthreads();

    bf16x16 a[2], b[2];
#pragma unroll
    for (int i = 0; i < 2; i++) a[i] = frag_ld(&As[wm * 32 + i * 16 + ln][0], kh);
#pragma unroll
    for (int j = 0; j < 2; j++) b[j] = frag_ld(&Bs[wn * 32 + j * 16 + ln][0], kh);
#pragma unroll
    for (int i = 0; i < 2; i++)
#pragma unroll
      for (int j = 0; j < 2; j++) acc[i][j] = wmma_bf16(a[i], b[j], acc[i][j]);
    __syncthreads();
  }

  // Epilogue. region/bb derived from block scalars only -> SCALAR branches
  // (a 64-col tile never crosses the q/k/v boundary; a 128-row tile never
  // crosses the batch boundary).
  const int region = nB >> 10;           // 0=q 1=k 2=v, uniform
  const int bb = mB >> 11;               // batch, uniform
  const int s0r = mB & (S_DIM - 1);
#pragma unroll
  for (int i = 0; i < 2; i++) {
#pragma unroll
    for (int j = 0; j < 2; j++) {
      int colE = ((nB + wn * 32 + j * 16) & 1023) + ln;  // e = col % 1024
      int h = colE >> 6, d = colE & 63;
      float bv = bias[nB + wn * 32 + j * 16 + ln];
      if (region == 2) {
        // V transposed [B,H,D,S]: lane's 8 elems are consecutive s -> b128
        int s = s0r + wm * 32 + i * 16 + 8 * kh;
        u16x8 pk;
#pragma unroll
        for (int v = 0; v < 8; ++v) pk[v] = f2bf(acc[i][j][v] + bv);
        *(u16x8*)(Vb + (((size_t)bb * H_DIM + h) * D_DIM + d) * S_DIM + s) = pk;
      } else {
        u16* dst = (region == 0) ? Qb : Kb;  // uniform select
#pragma unroll
        for (int v = 0; v < 8; ++v) {
          int s = s0r + wm * 32 + i * 16 + v + 8 * kh;
          dst[(((size_t)bb * H_DIM + h) * S_DIM + s) * D_DIM + d] =
              f2bf(acc[i][j][v] + bv);
        }
      }
    }
  }
}

// ---------------------------------------------------------------------------
// Kernel 2: flash attention. One WG per (bh, 128 query rows); each wave owns
// 16 rows. 64-key blocks, double-buffered async K/V staging.
// Grid: (2048/128, B*H); 256 threads.
// ---------------------------------------------------------------------------
__device__ __forceinline__ void stage_kv(const u16* Kg, const u16* Vg,
                                         u16 (*Kt)[72], u16 (*Vt)[72],
                                         int tid) {
  // K tile: 64 keys x 64 d ([s][d] rows); V tile: 64 d x 64 keys ([d][s] rows)
  // 512 x 16B chunks each; 2 per thread per tensor, all via ASYNCcnt.
#pragma unroll
  for (int it = 0; it < 2; ++it) {
    int slot = it * 256 + tid;
    int r = slot >> 3, c = slot & 7;
    async_b128(&Kt[r][c * 8], Kg + (size_t)r * D_DIM + c * 8);
    async_b128(&Vt[r][c * 8], Vg + (size_t)r * S_DIM + c * 8);
  }
}

__global__ __launch_bounds__(256) void k_attn(
    const u16* __restrict__ Qb, const u16* __restrict__ Kb,
    const u16* __restrict__ Vb, u16* __restrict__ Ob) {
  // stride 72 u16 = 144B: 16B-aligned, 36-DWORD rows -> 16 distinct banks
  __shared__ u16 Ks[2][64][72];
  __shared__ u16 Vs[2][64][72];
  __shared__ u16 Ps[8][16][72];  // per-wave P bounce (C-layout -> A-layout)

  const int tid = threadIdx.x, lane = tid & 31, wid = tid >> 5;
  const int kh = lane >> 4, ln = lane & 15;
  const int bh = blockIdx.y;
  const int bb = bh >> 4, h = bh & 15;
  const size_t baseK = (size_t)bh * S_DIM * D_DIM;  // Q,K: [s][d]
  const size_t baseV = (size_t)bh * D_DIM * S_DIM;  // V:   [d][s]
  const int q0 = blockIdx.x * 128 + wid * 16;

  // Q fragments resident in VGPRs (D=64 -> two K=32 chunks)
  const u16* qrow = Qb + baseK + (size_t)(q0 + ln) * D_DIM;
  bf16x16 aq0 = frag_ld(qrow, kh);
  bf16x16 aq1 = frag_ld(qrow + 32, kh);

  f32x8 o[4];
#pragma unroll
  for (int t = 0; t < 4; t++)
#pragma unroll
    for (int v = 0; v < 8; v++) o[t][v] = 0.f;
  float m[8], lsum[8];
#pragma unroll
  for (int v = 0; v < 8; v++) { m[v] = -1e30f; lsum[v] = 0.f; }
  const float scale = 0.125f;  // 1/sqrt(64)

  stage_kv(Kb + baseK, Vb + baseV, Ks[0], Vs[0], tid);  // prologue: block 0

  const int NBLK = S_DIM / 64;
  for (int i = 0; i < NBLK; ++i) {
    const int buf = i & 1;
    wait_async0();     // our async copies for `buf` are done
    __syncthreads();   // ... and everyone else's too
    if (i + 1 < NBLK)  // prefetch next block into the other buffer
      stage_kv(Kb + baseK + (size_t)(i + 1) * 64 * D_DIM,
               Vb + baseV + (i + 1) * 64, Ks[buf ^ 1], Vs[buf ^ 1], tid);

    // scores: four 16-key tiles, K-dim = D = 64 (2 WMMAs each)
    f32x8 sc[4];
#pragma unroll
    for (int t = 0; t < 4; t++) {
      const u16* kr = &Ks[buf][t * 16 + ln][0];
      f32x8 c;
#pragma unroll
      for (int v = 0; v < 8; v++) c[v] = 0.f;
      c = wmma_bf16(aq0, frag_ld(kr, kh), c);
      c = wmma_bf16(aq1, frag_ld(kr + 32, kh), c);
      sc[t] = c;
    }

    // online softmax over 64 keys; element v tracks query row q0 + v + 8*kh
    float alpha[8];
#pragma unroll
    for (int v = 0; v < 8; v++) {
      float s0 = sc[0][v] * scale, s1 = sc[1][v] * scale;
      float s2 = sc[2][v] * scale, s3 = sc[3][v] * scale;
      float mx = fmaxf(fmaxf(s0, s1), fmaxf(s2, s3));
#pragma unroll
      for (int off = 1; off < 16; off <<= 1) mx = fmaxf(mx, __shfl_xor(mx, off, 16));
      float mn = fmaxf(m[v], mx);
      alpha[v] = __expf(m[v] - mn);
      m[v] = mn;
      float p0 = __expf(s0 - mn), p1 = __expf(s1 - mn);
      float p2 = __expf(s2 - mn), p3 = __expf(s3 - mn);
      float ps = (p0 + p1) + (p2 + p3);
#pragma unroll
      for (int off = 1; off < 16; off <<= 1) ps += __shfl_xor(ps, off, 16);
      lsum[v] = lsum[v] * alpha[v] + ps;
      int pr = v + 8 * kh;  // C layout -> LDS bounce in A layout
      Ps[wid][pr][ln] = f2bf(p0);
      Ps[wid][pr][16 + ln] = f2bf(p1);
      Ps[wid][pr][32 + ln] = f2bf(p2);
      Ps[wid][pr][48 + ln] = f2bf(p3);
    }
#pragma unroll
    for (int t = 0; t < 4; t++)
#pragma unroll
      for (int v = 0; v < 8; v++) o[t][v] *= alpha[v];

    asm volatile("s_wait_dscnt 0x0" ::: "memory");   // wave-local LDS RAW
    bf16x16 ap0 = frag_ld(&Ps[wid][ln][0], kh);      // P keys 0..31
    bf16x16 ap1 = frag_ld(&Ps[wid][ln][0] + 32, kh); // P keys 32..63

    // O += P @ V : V tile rows are d (pre-transposed), cols are keys
#pragma unroll
    for (int t = 0; t < 4; t++) {
      const u16* vr = &Vs[buf][t * 16 + ln][0];
      o[t] = wmma_bf16(ap0, frag_ld(vr, kh), o[t]);
      o[t] = wmma_bf16(ap1, frag_ld(vr + 32, kh), o[t]);
    }
    // next iteration's wait_async + barrier fences buffer reuse
  }

  // finalize: O /= l, write bf16 attn output [B*S, E]
#pragma unroll
  for (int v = 0; v < 8; v++) lsum[v] = 1.f / lsum[v];
#pragma unroll
  for (int t = 0; t < 4; t++) {
#pragma unroll
    for (int v = 0; v < 8; v++) {
      int s = q0 + v + 8 * kh;
      Ob[(size_t)(bb * S_DIM + s) * E_DIM + h * D_DIM + t * 16 + ln] =
          f2bf(o[t][v] * lsum[v]);
    }
  }
}

// ---------------------------------------------------------------------------
// Kernel 3: out = Attn(bf16) @ Wout^T + bout (fp32). A tile staged with
// async-to-LDS (already bf16); B converted fp32->bf16 through VGPRs.
// Grid: (4096/128, 1024/64); 256 threads.
// ---------------------------------------------------------------------------
__global__ __launch_bounds__(256) void k_outproj(
    const u16* __restrict__ A, const float* __restrict__ W,
    const float* __restrict__ bias, float* __restrict__ out) {
  __shared__ u16 As[128][40];
  __shared__ u16 Bs[64][40];

  const int tid = threadIdx.x, lane = tid & 31, wid = tid >> 5;
  const int wm = wid & 3, wn = wid >> 2;
  const int kh = lane >> 4, ln = lane & 15;
  const int mB = blockIdx.x * 128, nB = blockIdx.y * 64;

  f32x8 acc[2][2];
#pragma unroll
  for (int i = 0; i < 2; i++)
#pragma unroll
    for (int j = 0; j < 2; j++)
#pragma unroll
      for (int v = 0; v < 8; v++) acc[i][j][v] = 0.f;

  for (int k0 = 0; k0 < E_DIM; k0 += 32) {
#pragma unroll
    for (int it = 0; it < 2; ++it) {  // A: 128x32 bf16, async 16B chunks
      int slot = it * 256 + tid;      // 512 chunks: 128 rows x 4
      int r = slot >> 2, c = slot & 3;
      async_b128(&As[r][c * 8], A + (size_t)(mB + r) * E_DIM + k0 + c * 8);
    }
#pragma unroll
    for (int it = 0; it < 2; ++it) {  // B: Wout fp32 -> bf16
      int slot = it * 256 + tid;
      int r = slot >> 3, c4 = slot & 7;
      f32x4 wv = *(const f32x4*)(W + (size_t)(nB + r) * E_DIM + k0 + c4 * 4);
      u16x4 bv;
      bv[0] = f2bf(wv[0]); bv[1] = f2bf(wv[1]);
      bv[2] = f2bf(wv[2]); bv[3] = f2bf(wv[3]);
      *(u16x4*)(&Bs[r][c4 * 4]) = bv;
    }
    wait_async0();
    __syncthreads();

    bf16x16 a[2], b[2];
#pragma unroll
    for (int i = 0; i < 2; i++) a[i] = frag_ld(&As[wm * 32 + i * 16 + ln][0], kh);
#pragma unroll
    for (int j = 0; j < 2; j++) b[j] = frag_ld(&Bs[wn * 32 + j * 16 + ln][0], kh);
#pragma unroll
    for (int i = 0; i < 2; i++)
#pragma unroll
      for (int j = 0; j < 2; j++) acc[i][j] = wmma_bf16(a[i], b[j], acc[i][j]);
    __syncthreads();
  }

#pragma unroll
  for (int i = 0; i < 2; i++) {
#pragma unroll
    for (int j = 0; j < 2; j++) {
      int col = nB + wn * 32 + j * 16 + ln;
      float bv = bias[col];
#pragma unroll
      for (int v = 0; v < 8; ++v) {
        int row = mB + wm * 32 + i * 16 + v + 8 * kh;
        out[(size_t)row * E_DIM + col] = acc[i][j][v] + bv;
      }
    }
  }
}

// ---------------------------------------------------------------------------
// Host launcher. Workspace (u16 elems): Qb 8MB | Kb 8MB | Vb(pre-T) 8MB |
// Ob(attn out) 8MB => 32MB total.
// ---------------------------------------------------------------------------
extern "C" void kernel_launch(void* const* d_in, const int* in_sizes, int n_in,
                              void* d_out, int out_size, void* d_ws,
                              size_t ws_size, hipStream_t stream) {
  (void)in_sizes; (void)n_in; (void)out_size; (void)ws_size;
  const float* query = (const float*)d_in[0];
  // d_in[1] (key), d_in[2] (value) unused: reference projects q,k,v from query
  const float* Wqkv = (const float*)d_in[3];
  const float* bqkv = (const float*)d_in[4];
  const float* Wout = (const float*)d_in[5];
  const float* bout = (const float*)d_in[6];

  const size_t headElems = (size_t)B_DIM * H_DIM * S_DIM * D_DIM;  // 4M
  u16* Qb = (u16*)d_ws;
  u16* Kb = Qb + headElems;
  u16* Vb = Kb + headElems;  // [B,H,D,S]
  u16* Ob = Vb + headElems;  // bf16 [B*S, E]

  k_inproj<<<dim3((B_DIM * S_DIM) / 128, (3 * E_DIM) / 64), 256, 0, stream>>>(
      query, Wqkv, bqkv, Qb, Kb, Vb);
  k_attn<<<dim3(S_DIM / 128, B_DIM * H_DIM), 256, 0, stream>>>(Qb, Kb, Vb, Ob);
  k_outproj<<<dim3((B_DIM * S_DIM) / 128, E_DIM / 64), 256, 0, stream>>>(
      Ob, Wout, bout, (float*)d_out);
}
